// OptimizedMoE_73375221284965
// MI455X (gfx1250) — compile-verified, matrix operand order checked by reading, not run on hardware
//
#include <hip/hip_runtime.h>
#include <hip/hip_bf16.h>
#include <math.h>

// Problem constants (match reference)
#define TOK  8192   // B*N tokens
#define CDIM 768
#define HDIM 1536
#define NEXP 8
#define CAP  8192   // per-expert token-list capacity (worst case for top-2)
#define MT   32     // tokens per workgroup tile (2 WMMA m-subtiles)

typedef __attribute__((ext_vector_type(16))) __bf16 v16bf;
typedef __attribute__((ext_vector_type(8)))  float  v8f;

union B16Frag { uint4 u[2]; v16bf v; };

__device__ __forceinline__ unsigned short f32_to_bf16_rte(float f) {
  unsigned int u = __float_as_uint(f);
  u += 0x7fffu + ((u >> 16) & 1u);
  return (unsigned short)(u >> 16);
}

// ---------------------------------------------------------------------------
// 0) zero output + expert counters
__global__ void k_init(float* __restrict__ out, int outN, int* __restrict__ cnt) {
  int i = blockIdx.x * blockDim.x + threadIdx.x;
  if (i < outN) out[i] = 0.0f;
  if (i < NEXP) cnt[i] = 0;
}

// 1) x (fp32) -> bf16
__global__ void k_cvt_x(const float* __restrict__ x, unsigned short* __restrict__ xb) {
  int i = blockIdx.x * blockDim.x + threadIdx.x;
  if (i < TOK * CDIM) xb[i] = f32_to_bf16_rte(x[i]);
}

// 2) W1 [E][C][H] -> W1T [E][H][C] bf16  (WMMA B operand: per-lane K-contiguous)
__global__ void k_tr_w1(const float* __restrict__ w1, unsigned short* __restrict__ w1t) {
  int i = blockIdx.x * blockDim.x + threadIdx.x;
  if (i >= NEXP * HDIM * CDIM) return;
  int c = i % CDIM;
  int t = i / CDIM;
  int h = t % HDIM;
  int e = t / HDIM;
  w1t[i] = f32_to_bf16_rte(w1[((size_t)e * CDIM + c) * HDIM + h]);
}

// 3) W2 [E][H][C] -> W2T [E][C][H] bf16
__global__ void k_tr_w2(const float* __restrict__ w2, unsigned short* __restrict__ w2t) {
  int i = blockIdx.x * blockDim.x + threadIdx.x;
  if (i >= NEXP * CDIM * HDIM) return;
  int h = i % HDIM;
  int t = i / HDIM;
  int c = t % CDIM;
  int e = t / CDIM;
  w2t[i] = f32_to_bf16_rte(w2[((size_t)e * HDIM + h) * CDIM + c]);
}

// 4) gating: logits -> softmax -> top2 -> renorm -> scatter into expert lists
__global__ void k_gate(const float* __restrict__ x, const float* __restrict__ Wg,
                       const float* __restrict__ bg,
                       int* __restrict__ lists, float* __restrict__ gws,
                       int* __restrict__ cnt) {
  int t = blockIdx.x * blockDim.x + threadIdx.x;
  if (t >= TOK) return;
  const float* xr = x + (size_t)t * CDIM;
  float acc[NEXP];
#pragma unroll
  for (int e = 0; e < NEXP; ++e) acc[e] = bg[e];
  for (int k = 0; k < CDIM; ++k) {
    float xv = xr[k];
    const float* wr = Wg + (size_t)k * NEXP;
#pragma unroll
    for (int e = 0; e < NEXP; ++e) acc[e] = fmaf(xv, wr[e], acc[e]);
  }
  float mx = acc[0];
#pragma unroll
  for (int e = 1; e < NEXP; ++e) mx = fmaxf(mx, acc[e]);
  float p[NEXP];
#pragma unroll
  for (int e = 0; e < NEXP; ++e) p[e] = __expf(acc[e] - mx);
  // top-2 (full softmax denominator cancels under top-2 renormalization)
  int i1 = 0;
#pragma unroll
  for (int e = 1; e < NEXP; ++e) if (p[e] > p[i1]) i1 = e;
  int i2 = (i1 == 0) ? 1 : 0;
#pragma unroll
  for (int e = 0; e < NEXP; ++e) if (e != i1 && p[e] > p[i2]) i2 = e;
  float g1 = p[i1], g2 = p[i2];
  float r = 1.0f / (g1 + g2);
  g1 *= r; g2 *= r;
  int p1 = atomicAdd(&cnt[i1], 1);
  lists[i1 * CAP + p1] = t; gws[i1 * CAP + p1] = g1;
  int p2 = atomicAdd(&cnt[i2], 1);
  lists[i2 * CAP + p2] = t; gws[i2 * CAP + p2] = g2;
}

// 5) fused expert kernel: one workgroup per (expert, 32-token tile).
//    x tile in LDS -> GEMM1 (bf16 WMMA, B frag reused for 2 m-subtiles,
//    ping-pong B buffers) -> exact erf GELU -> h in LDS -> GEMM2 ->
//    gated atomic-add combine.
__global__ __launch_bounds__(256)
void moe_expert_tile(const unsigned short* __restrict__ xb,   // [TOK][C] bf16
                     const unsigned short* __restrict__ w1t,  // [E][H][C] bf16
                     const unsigned short* __restrict__ w2t,  // [E][C][H] bf16
                     const float* __restrict__ b1,            // [E][H]
                     const float* __restrict__ b2,            // [E][C]
                     const int* __restrict__ lists,           // [E][CAP]
                     const float* __restrict__ gws,           // [E][CAP]
                     const int* __restrict__ cnts,            // [E]
                     float* __restrict__ out)                 // [TOK][C]
{
  const int e  = blockIdx.y;
  const int m0 = blockIdx.x * MT;
  const int cnt = cnts[e];
  if (m0 >= cnt) return;                      // uniform early exit (padded grid)
  const int valid = min(MT, cnt - m0);

  __shared__ int   s_tok[MT];
  __shared__ float s_gw[MT];
  __shared__ __align__(16) unsigned short s_x[MT * CDIM];   // 48 KB x tile (bf16)
  __shared__ __align__(16) unsigned short s_h[MT * HDIM];   // 96 KB hidden tile (bf16)

  const int tid = threadIdx.x;
  if (tid < MT) {
    int r = m0 + tid;
    if (r < cnt) { s_tok[tid] = lists[e * CAP + r]; s_gw[tid] = gws[e * CAP + r]; }
    else         { s_tok[tid] = lists[e * CAP];     s_gw[tid] = 0.0f; }
  }
  __syncthreads();

  // Stage gathered x rows into LDS (16B chunks, coalesced within a row)
  {
    const int cpr = CDIM / 8;                 // uint4 chunks per row (96)
    for (int c = tid; c < MT * cpr; c += 256) {
      int row = c / cpr;
      int col = (c - row * cpr) * 8;
      uint4 v = *(const uint4*)(xb + (size_t)s_tok[row] * CDIM + col);
      *(uint4*)(&s_x[row * CDIM + col]) = v;
    }
  }
  __syncthreads();

  const int wave  = tid >> 5;        // 0..7 (wave32)
  const int lane  = tid & 31;
  const int lm    = lane & 15;       // row (A) / col (B,D) within 16-tile
  const int lk    = (lane >> 4) * 8; // K sub-chunk select per ISA layout
  const int mbase = (lane >> 4) * 8; // D: M = mbase + vgpr

  // ---------------- GEMM1: h = gelu(x_tile @ W1_e + b1) ----------------
  const unsigned short* w1e = w1t + (size_t)e * HDIM * CDIM;
  const unsigned short* a0  = s_x + (size_t)lm * CDIM;          // m-subtile 0 row
  const unsigned short* a1  = s_x + (size_t)(16 + lm) * CDIM;   // m-subtile 1 row
  for (int nt = wave * 12; nt < wave * 12 + 12; ++nt) {         // 96 N-tiles / 8 waves
    const int ncol = nt * 16 + lm;
    const unsigned short* brow = w1e + (size_t)ncol * CDIM;
    __builtin_prefetch(brow + CDIM * 16, 0, 1);
    v8f acc0 = {}, acc1 = {};
    B16Frag bb[2];                            // ping-pong: overlapping live ranges,
    bb[0].u[0] = *(const uint4*)(brow + lk);  // no copies -> two physical reg pairs
    bb[0].u[1] = *(const uint4*)(brow + lk + 16);
#pragma unroll
    for (int i = 0; i < CDIM / 32; ++i) {
      const int ks = i * 32;
      if (i + 1 < CDIM / 32) {
        bb[(i + 1) & 1].u[0] = *(const uint4*)(brow + ks + 32 + lk);
        bb[(i + 1) & 1].u[1] = *(const uint4*)(brow + ks + 32 + lk + 16);
      }
      B16Frag af0, af1;
      af0.u[0] = *(const uint4*)(a0 + ks + lk);
      af0.u[1] = *(const uint4*)(a0 + ks + lk + 16);
      af1.u[0] = *(const uint4*)(a1 + ks + lk);
      af1.u[1] = *(const uint4*)(a1 + ks + lk + 16);
      acc0 = __builtin_amdgcn_wmma_f32_16x16x32_bf16(false, af0.v, false, bb[i & 1].v,
                                                     (short)0, acc0, false, false);
      acc1 = __builtin_amdgcn_wmma_f32_16x16x32_bf16(false, af1.v, false, bb[i & 1].v,
                                                     (short)0, acc1, false, false);
    }
    const float bias = b1[e * HDIM + ncol];
#pragma unroll
    for (int r = 0; r < 8; ++r) {
      float v0 = acc0[r] + bias;
      float v1 = acc1[r] + bias;
      float g0 = 0.5f * v0 * (1.0f + erff(v0 * 0.70710678118654752f)); // exact erf GELU
      float g1 = 0.5f * v1 * (1.0f + erff(v1 * 0.70710678118654752f));
      s_h[(mbase + r) * HDIM + ncol]        = f32_to_bf16_rte(g0);
      s_h[(16 + mbase + r) * HDIM + ncol]   = f32_to_bf16_rte(g1);
    }
  }
  __syncthreads();

  // ---------------- GEMM2: y = h @ W2_e + b2 ; out += gate * y ----------------
  const unsigned short* w2e = w2t + (size_t)e * CDIM * HDIM;
  const unsigned short* h0  = s_h + (size_t)lm * HDIM;
  const unsigned short* h1  = s_h + (size_t)(16 + lm) * HDIM;
  for (int nt = wave * 6; nt < wave * 6 + 6; ++nt) {            // 48 N-tiles / 8 waves
    const int ncol = nt * 16 + lm;
    const unsigned short* brow = w2e + (size_t)ncol * HDIM;
    __builtin_prefetch(brow + HDIM * 16, 0, 1);
    v8f acc0 = {}, acc1 = {};
    B16Frag bb[2];
    bb[0].u[0] = *(const uint4*)(brow + lk);
    bb[0].u[1] = *(const uint4*)(brow + lk + 16);
#pragma unroll
    for (int i = 0; i < HDIM / 32; ++i) {
      const int ks = i * 32;
      if (i + 1 < HDIM / 32) {
        bb[(i + 1) & 1].u[0] = *(const uint4*)(brow + ks + 32 + lk);
        bb[(i + 1) & 1].u[1] = *(const uint4*)(brow + ks + 32 + lk + 16);
      }
      B16Frag af0, af1;
      af0.u[0] = *(const uint4*)(h0 + ks + lk);
      af0.u[1] = *(const uint4*)(h0 + ks + lk + 16);
      af1.u[0] = *(const uint4*)(h1 + ks + lk);
      af1.u[1] = *(const uint4*)(h1 + ks + lk + 16);
      acc0 = __builtin_amdgcn_wmma_f32_16x16x32_bf16(false, af0.v, false, bb[i & 1].v,
                                                     (short)0, acc0, false, false);
      acc1 = __builtin_amdgcn_wmma_f32_16x16x32_bf16(false, af1.v, false, bb[i & 1].v,
                                                     (short)0, acc1, false, false);
    }
    const float bias = b2[e * CDIM + ncol];
#pragma unroll
    for (int r = 0; r < 8; ++r) {
      int ma = mbase + r;
      int mb = 16 + mbase + r;
      if (ma < valid) {
        float v = (acc0[r] + bias) * s_gw[ma];
        atomicAdd(&out[(size_t)s_tok[ma] * CDIM + ncol], v);  // <=2-way contention
      }
      if (mb < valid) {
        float v = (acc1[r] + bias) * s_gw[mb];
        atomicAdd(&out[(size_t)s_tok[mb] * CDIM + ncol], v);
      }
    }
  }
}

// ---------------------------------------------------------------------------
extern "C" void kernel_launch(void* const* d_in, const int* in_sizes, int n_in,
                              void* d_out, int out_size, void* d_ws, size_t ws_size,
                              hipStream_t stream) {
  const float* x  = (const float*)d_in[0];
  const float* Wg = (const float*)d_in[1];
  const float* bg = (const float*)d_in[2];
  const float* W1 = (const float*)d_in[3];
  const float* b1 = (const float*)d_in[4];
  const float* W2 = (const float*)d_in[5];
  const float* b2 = (const float*)d_in[6];
  float* out = (float*)d_out;

  char* ws = (char*)d_ws;
  size_t off = 0;
  unsigned short* xb  = (unsigned short*)(ws + off); off += (size_t)TOK * CDIM * 2;           // 12.6 MB
  unsigned short* w1t = (unsigned short*)(ws + off); off += (size_t)NEXP * HDIM * CDIM * 2;   // 18.9 MB
  unsigned short* w2t = (unsigned short*)(ws + off); off += (size_t)NEXP * CDIM * HDIM * 2;   // 18.9 MB
  int*   lists = (int*)  (ws + off); off += (size_t)NEXP * CAP * 4;                           // 256 KB
  float* gws   = (float*)(ws + off); off += (size_t)NEXP * CAP * 4;                           // 256 KB
  int*   cnt   = (int*)  (ws + off); off += 256;

  const int outN = TOK * CDIM;
  k_init <<<(outN + 255) / 256, 256, 0, stream>>>(out, outN, cnt);
  k_cvt_x<<<(TOK * CDIM + 255) / 256, 256, 0, stream>>>(x, xb);
  k_tr_w1<<<(NEXP * HDIM * CDIM + 255) / 256, 256, 0, stream>>>(W1, w1t);
  k_tr_w2<<<(NEXP * CDIM * HDIM + 255) / 256, 256, 0, stream>>>(W2, w2t);
  k_gate <<<(TOK + 255) / 256, 256, 0, stream>>>(x, Wg, bg, lists, gws, cnt);

  dim3 grid(CAP / MT, NEXP);   // static worst-case grid; inactive tiles early-exit
  moe_expert_tile<<<grid, 256, 0, stream>>>(xb, w1t, w2t, b1, b2, lists, gws, cnt, out);
}
